// MoE_34754875359705
// MI455X (gfx1250) — matrix operand order repaired and checked by reading, not compiled, not run on hardware
//
#include <hip/hip_runtime.h>
#include <hip/hip_bf16.h>
#include <math.h>

// ---- problem constants (from reference setup_inputs) ----
#define TB 2
#define TL 2048
#define TD 2048      // model dim D
#define TE 8         // experts
#define TH 2048      // expert hidden H
#define TG 256       // gate hidden G
#define TT (TB*TL)   // tokens T = 4096
#define TK 2048      // shared K extent of all big GEMMs (D and H)
#define EPSL 1e-9f

#define USE_ASYNC_LDS 1

typedef __bf16 v16bf __attribute__((ext_vector_type(16)));
typedef __bf16 v8bf  __attribute__((ext_vector_type(8)));
typedef float  v8f   __attribute__((ext_vector_type(8)));

static __device__ __forceinline__ __bf16 f2bf(float f) { return (__bf16)f; }
static __device__ __forceinline__ float bf2f(__bf16 b) {
  unsigned short h = __builtin_bit_cast(unsigned short, b);
  unsigned u = ((unsigned)h) << 16;
  return __builtin_bit_cast(float, u);
}
static __device__ __forceinline__ float gelu_tanh(float v) {
  const float c0 = 0.7978845608028654f, c1 = 0.044715f;
  float u = c0 * (v + c1 * v * v * v);
  return 0.5f * v * (1.0f + tanhf(u));
}
static __device__ __forceinline__ v16bf cat8(v8bf a, v8bf b) {
  return __builtin_shufflevector(a, b, 0,1,2,3,4,5,6,7,8,9,10,11,12,13,14,15);
}
static __device__ __forceinline__ v8f wmma_bf16(v16bf a, v16bf b, v8f c) {
  return __builtin_amdgcn_wmma_f32_16x16x32_bf16(false, a, false, b, (short)0, c, false, false);
}

// 16-byte global -> LDS copy. Async path uses the CDNA5 async-DMA instruction
// (tracked by ASYNCcnt, no VGPR round-trip); generic LDS pointers carry the LDS
// byte offset in their low 32 bits (flat-aperture rule, ISA 10.2).
static __device__ __forceinline__ void g2lds16(__bf16* lds_dst, const __bf16* gsrc) {
#if USE_ASYNC_LDS
  unsigned lds_off = (unsigned)(uintptr_t)lds_dst;
  unsigned long long ga = (unsigned long long)(uintptr_t)gsrc;
  asm volatile("global_load_async_to_lds_b128 %0, %1, off"
               :: "v"(lds_off), "v"(ga) : "memory");
#else
  *(uint4*)lds_dst = *(const uint4*)gsrc;
#endif
}
static __device__ __forceinline__ void g2lds_wait() {
#if USE_ASYNC_LDS
  asm volatile("s_wait_asynccnt 0x0" ::: "memory");
#endif
}

// ============ one-time fp32 -> bf16 convert (row-major, elementwise) ============
__global__ __launch_bounds__(256) void cvt_kernel(const float* __restrict__ src,
                                                  __bf16* __restrict__ dst) {
  size_t i = ((size_t)blockIdx.x * 256 + threadIdx.x) * 8;
  const float4* s = (const float4*)(src + i);
  float4 f0 = s[0], f1 = s[1];
  __bf16* d = dst + i;
  d[0] = f2bf(f0.x); d[1] = f2bf(f0.y); d[2] = f2bf(f0.z); d[3] = f2bf(f0.w);
  d[4] = f2bf(f1.x); d[5] = f2bf(f1.y); d[6] = f2bf(f1.z); d[7] = f2bf(f1.w);
}

// ====== one-time fp32 [K][N] -> bf16 transposed [N][K] (64x64 LDS tiles) ======
__global__ __launch_bounds__(256) void cvt_transpose_kernel(const float* __restrict__ src,
                                                            __bf16* __restrict__ dst,
                                                            int K, int N) {
  __shared__ __bf16 tile[64][72];
  const float* s = src + (size_t)blockIdx.z * K * N;
  __bf16* d = dst + (size_t)blockIdx.z * K * N;
  const int k0 = blockIdx.y * 64, n0 = blockIdx.x * 64;
  const int r = threadIdx.x >> 2, sg = threadIdx.x & 3;
  const float4* sp = (const float4*)(s + (size_t)(k0 + r) * N + n0 + sg * 16);
  float4 f0 = sp[0], f1 = sp[1], f2 = sp[2], f3 = sp[3];
  __bf16* t = &tile[r][sg * 16];
  t[0]=f2bf(f0.x);  t[1]=f2bf(f0.y);  t[2]=f2bf(f0.z);  t[3]=f2bf(f0.w);
  t[4]=f2bf(f1.x);  t[5]=f2bf(f1.y);  t[6]=f2bf(f1.z);  t[7]=f2bf(f1.w);
  t[8]=f2bf(f2.x);  t[9]=f2bf(f2.y);  t[10]=f2bf(f2.z); t[11]=f2bf(f2.w);
  t[12]=f2bf(f3.x); t[13]=f2bf(f3.y); t[14]=f2bf(f3.z); t[15]=f2bf(f3.w);
  __syncthreads();
  __bf16* dp = d + (size_t)(n0 + r) * K + k0 + sg * 16;
#pragma unroll
  for (int j = 0; j < 16; ++j) dp[j] = tile[sg * 16 + j][r];
}

// ================= unified 128x128-tile WMMA GEMM =================
// MODE 0: A1 = bf16(gelu(xb @ wg1T^T))          rows = tokens directly
// MODE 1: hb = bf16(gate * gelu(xb_gath @ w1T^T)) rows gathered via token_map
// MODE 2: y[token] += hb @ w2T^T                 rows = expert slots, scatter-add
template <int MODE>
__global__ __launch_bounds__(256) void gemm_kernel(const __bf16* __restrict__ Abase,
                                                   const __bf16* __restrict__ BTbase,
                                                   const int* __restrict__ token_map,
                                                   const float* __restrict__ gate_map,
                                                   const int* __restrict__ counts,
                                                   const int* __restrict__ offsets,
                                                   __bf16* __restrict__ out_bf,
                                                   float* __restrict__ out_f) {
  int cnt = TT, base = 0;
  const __bf16* BT = BTbase;
  if (MODE >= 1) {
    const int e = blockIdx.z;
    cnt = counts[e];
    if ((int)blockIdx.y * 128 >= cnt) return;
    base = offsets[e];
    BT = BTbase + (size_t)e * TK * TK;
  }
  const int n0 = blockIdx.x * 128;
  const int m0 = blockIdx.y * 128;
  const int tid = threadIdx.x, lane = tid & 31, wave = tid >> 5;
  const int wm = (wave & 3) * 32, wn = (wave >> 2) * 64;
  const int arow = tid >> 1, ahalf = tid & 1;

  __shared__ __align__(16) __bf16 As[2][128][40];
  __shared__ __align__(16) __bf16 Bt[2][128][40];

  // per-thread source row pointers (gather resolved once)
  int arow_idx;
  if (MODE == 0) {
    arow_idx = m0 + arow;
  } else if (MODE == 1) {
    int slot = m0 + arow;
    arow_idx = (slot < cnt) ? token_map[base + slot] : 0;
  } else {
    int slot = m0 + arow;
    arow_idx = base + ((slot < cnt) ? slot : (cnt - 1));
  }
  const __bf16* aptr = Abase + (size_t)arow_idx * TK + ahalf * 16;
  const __bf16* bptr = BT + (size_t)(n0 + arow) * TK + ahalf * 16;
  __bf16* asdst0 = &As[0][arow][ahalf * 16];
  __bf16* asdst1 = &As[1][arow][ahalf * 16];
  __bf16* btdst0 = &Bt[0][arow][ahalf * 16];
  __bf16* btdst1 = &Bt[1][arow][ahalf * 16];

  v8f acc[2][4] = {};

  // prologue: stage k-tile 0 into buffer 0
  g2lds16(asdst0, aptr);       g2lds16(asdst0 + 8, aptr + 8);
  g2lds16(btdst0, bptr);       g2lds16(btdst0 + 8, bptr + 8);

  const int mlb = wm + (lane & 15), nlb = wn + (lane & 15), h = lane >> 4;
  int buf = 0;
  for (int k0 = 0; k0 < TK; k0 += 32) {
    g2lds_wait();          // my async writes into As[buf]/Bt[buf] are done
    __syncthreads();       // everyone's are done -> safe to read buf
    if (k0 + 32 < TK) {    // prefetch next k-tile into the other buffer
      const __bf16* an = aptr + k0 + 32;
      const __bf16* bn = bptr + k0 + 32;
      __bf16* ad = buf ? asdst0 : asdst1;
      __bf16* bd = buf ? btdst0 : btdst1;
      g2lds16(ad, an); g2lds16(ad + 8, an + 8);
      g2lds16(bd, bn); g2lds16(bd + 8, bn + 8);
    }
    v16bf af[2], bfr[4];
#pragma unroll
    for (int mi = 0; mi < 2; ++mi) {
      v8bf c0 = *(const v8bf*)&As[buf][mlb + mi * 16][h * 8];
      v8bf c1 = *(const v8bf*)&As[buf][mlb + mi * 16][16 + h * 8];
      af[mi] = cat8(c0, c1);
    }
#pragma unroll
    for (int ni = 0; ni < 4; ++ni) {
      v8bf c0 = *(const v8bf*)&Bt[buf][nlb + ni * 16][h * 8];
      v8bf c1 = *(const v8bf*)&Bt[buf][nlb + ni * 16][16 + h * 8];
      bfr[ni] = cat8(c0, c1);
    }
#pragma unroll
    for (int mi = 0; mi < 2; ++mi)
#pragma unroll
      for (int ni = 0; ni < 4; ++ni)
        acc[mi][ni] = wmma_bf16(af[mi], bfr[ni], acc[mi][ni]);
    buf ^= 1;
  }

  // ---- epilogue ----
#pragma unroll
  for (int mi = 0; mi < 2; ++mi)
#pragma unroll
    for (int i = 0; i < 8; ++i) {
      const int mrow = wm + mi * 16 + (lane >> 4) * 8 + i;
      if (MODE == 0) {
        const int m = m0 + mrow;
#pragma unroll
        for (int ni = 0; ni < 4; ++ni) {
          int n = n0 + wn + ni * 16 + (lane & 15);
          out_bf[(size_t)m * TG + n] = f2bf(gelu_tanh(acc[mi][ni][i]));
        }
      } else if (MODE == 1) {
        const int slot = m0 + mrow;
        if (slot < cnt) {
          float g = gate_map[base + slot];
#pragma unroll
          for (int ni = 0; ni < 4; ++ni) {
            int n = n0 + wn + ni * 16 + (lane & 15);
            out_bf[(size_t)(base + slot) * TH + n] = f2bf(g * gelu_tanh(acc[mi][ni][i]));
          }
        }
      } else {
        const int slot = m0 + mrow;
        if (slot < cnt) {
          int token = token_map[base + slot];
#pragma unroll
          for (int ni = 0; ni < 4; ++ni) {
            int n = n0 + wn + ni * 16 + (lane & 15);
            atomicAdd(&out_f[(size_t)token * TD + n], acc[mi][ni][i]);
          }
        }
      }
    }
}

// ================= routing (logits, softmax, top-2, counts) =================
__global__ __launch_bounds__(256) void route_kernel(const __bf16* __restrict__ A1,
                                                    const float* __restrict__ wg2,
                                                    float* __restrict__ probs,
                                                    int* __restrict__ top_idx,
                                                    float* __restrict__ top_val,
                                                    int* __restrict__ counts) {
  const int wave = threadIdx.x >> 5, lane = threadIdx.x & 31;
  const int t = blockIdx.x * 8 + wave;
  float acc[TE];
#pragma unroll
  for (int e = 0; e < TE; ++e) acc[e] = 0.f;
  for (int g = lane; g < TG; g += 32) {
    float a = bf2f(A1[(size_t)t * TG + g]);
#pragma unroll
    for (int e = 0; e < TE; ++e) acc[e] += a * wg2[g * TE + e];
  }
#pragma unroll
  for (int off = 16; off > 0; off >>= 1)
#pragma unroll
    for (int e = 0; e < TE; ++e) acc[e] += __shfl_xor(acc[e], off, 32);
  if (lane == 0) {
    float mx = acc[0];
#pragma unroll
    for (int e = 1; e < TE; ++e) mx = fmaxf(mx, acc[e]);
    float p[TE], den = 0.f;
#pragma unroll
    for (int e = 0; e < TE; ++e) { p[e] = expf(acc[e] - mx); den += p[e]; }
    float inv = 1.f / den;
#pragma unroll
    for (int e = 0; e < TE; ++e) { p[e] *= inv; probs[(size_t)t * TE + e] = p[e]; }
    int i0 = 0;
#pragma unroll
    for (int e = 1; e < TE; ++e) if (p[e] > p[i0]) i0 = e;
    int i1 = (i0 == 0) ? 1 : 0;
#pragma unroll
    for (int e = 0; e < TE; ++e) if (e != i0 && p[e] > p[i1]) i1 = e;
    top_idx[t * 2 + 0] = i0; top_idx[t * 2 + 1] = i1;
    top_val[t * 2 + 0] = p[i0]; top_val[t * 2 + 1] = p[i1];
    atomicAdd(&counts[i0], 1);
    atomicAdd(&counts[i1], 1);
  }
}

// ================= exclusive prefix over E=8 =================
__global__ void prefix_kernel(const int* __restrict__ counts, int* __restrict__ offsets,
                              int* __restrict__ cursor) {
  if (threadIdx.x == 0) {
    int s = 0;
#pragma unroll
    for (int e = 0; e < TE; ++e) { offsets[e] = s; s += counts[e]; cursor[e] = 0; }
  }
}

// ================= scatter tokens into per-expert compacted lists =================
__global__ __launch_bounds__(256) void scatter_kernel(const int* __restrict__ top_idx,
                                                      const float* __restrict__ top_val,
                                                      const int* __restrict__ offsets,
                                                      int* __restrict__ cursor,
                                                      int* __restrict__ token_map,
                                                      float* __restrict__ gate_map) {
  const int t = blockIdx.x * 256 + threadIdx.x;
  if (t >= TT) return;
#pragma unroll
  for (int k = 0; k < 2; ++k) {
    int e = top_idx[t * 2 + k];
    int pos = offsets[e] + atomicAdd(&cursor[e], 1);
    token_map[pos] = t;
    gate_map[pos] = top_val[t * 2 + k];
  }
}

// ================= deterministic aux-loss reduction =================
__global__ __launch_bounds__(256) void loss_kernel(const float* __restrict__ probs,
                                                   float* __restrict__ out_loss) {
  __shared__ float red[256];
  __shared__ float fin[TE + 1];
  const int tid = threadIdx.x;
  float pm[TE];
#pragma unroll
  for (int e = 0; e < TE; ++e) pm[e] = 0.f;
  float hc = 0.f;
  for (int t = tid; t < TT; t += 256) {
#pragma unroll
    for (int e = 0; e < TE; ++e) {
      float p = probs[(size_t)t * TE + e];
      pm[e] += p;
      hc += p * logf(p + EPSL);
    }
  }
  for (int e = 0; e <= TE; ++e) {
    red[tid] = (e < TE) ? pm[e] : hc;
    __syncthreads();
    for (int s = 128; s > 0; s >>= 1) {
      if (tid < s) red[tid] += red[tid + s];
      __syncthreads();
    }
    if (tid == 0) fin[e] = red[0];
    __syncthreads();
  }
  if (tid == 0) {
    float loss = 0.f;
#pragma unroll
    for (int e = 0; e < TE; ++e) {
      float pmean = fin[e] / (float)TT;
      loss += pmean * logf(pmean + EPSL);
    }
    loss -= fin[TE] / (float)TT;
    *out_loss = loss;
  }
}

// ================= host-side orchestration =================
extern "C" void kernel_launch(void* const* d_in, const int* in_sizes, int n_in,
                              void* d_out, int out_size, void* d_ws, size_t ws_size,
                              hipStream_t stream) {
  (void)in_sizes; (void)n_in; (void)ws_size;
  const float* x   = (const float*)d_in[0];   // [T, D]
  const float* wg1 = (const float*)d_in[1];   // [D, G]
  const float* wg2 = (const float*)d_in[2];   // [G, E]
  const float* w1  = (const float*)d_in[3];   // [E, D, H]
  const float* w2  = (const float*)d_in[4];   // [E, H, D]
  float* y = (float*)d_out;                   // [B*L*D] + loss at [out_size-1]
  char* ws = (char*)d_ws;

  // ---- workspace layout ----
  const size_t oXb    = 0;                                 // T*D bf16   16 MB
  const size_t oWg1T  = oXb   + (size_t)TT * TD * 2;       // G*D bf16    1 MB
  const size_t oW1T   = oWg1T + (size_t)TG * TD * 2;       // E*H*D bf16 64 MB
  const size_t oW2T   = oW1T  + (size_t)TE * TD * TH * 2;  // E*D*H bf16 64 MB
  const size_t oA1    = oW2T  + (size_t)TE * TH * TD * 2;  // T*G bf16    2 MB
  const size_t oProbs = oA1   + (size_t)TT * TG * 2;       // T*E f32
  const size_t oTopIdx= oProbs + (size_t)TT * TE * 4;
  const size_t oTopVal= oTopIdx + (size_t)TT * 2 * 4;
  const size_t oTokMap= oTopVal + (size_t)TT * 2 * 4;
  const size_t oGateMap= oTokMap + (size_t)TT * 2 * 4;
  const size_t oMeta  = oGateMap + (size_t)TT * 2 * 4;     // counts/offsets/cursor
  const size_t oHb    = oMeta + 256;                       // 2T*H bf16  32 MB

  __bf16* xb      = (__bf16*)(ws + oXb);
  __bf16* wg1T    = (__bf16*)(ws + oWg1T);
  __bf16* w1T     = (__bf16*)(ws + oW1T);
  __bf16* w2T     = (__bf16*)(ws + oW2T);
  __bf16* A1      = (__bf16*)(ws + oA1);
  float*  probs   = (float*)(ws + oProbs);
  int*    top_idx = (int*)(ws + oTopIdx);
  float*  top_val = (float*)(ws + oTopVal);
  int*    tok_map = (int*)(ws + oTokMap);
  float*  gate_map= (float*)(ws + oGateMap);
  int*    counts  = (int*)(ws + oMeta);
  int*    offsets = counts + TE;
  int*    cursor  = counts + 2 * TE;
  __bf16* hb      = (__bf16*)(ws + oHb);

  hipMemsetAsync(d_out, 0, (size_t)out_size * sizeof(float), stream);
  hipMemsetAsync(ws + oMeta, 0, 256, stream);

  // one-time precision/layout conversion (bandwidth-bound)
  cvt_kernel<<<(TT * TD) / (256 * 8), 256, 0, stream>>>(x, xb);
  cvt_transpose_kernel<<<dim3(TG / 64, TD / 64, 1), 256, 0, stream>>>(wg1, wg1T, TD, TG);
  cvt_transpose_kernel<<<dim3(TH / 64, TD / 64, TE), 256, 0, stream>>>(w1, w1T, TD, TH);
  cvt_transpose_kernel<<<dim3(TD / 64, TH / 64, TE), 256, 0, stream>>>(w2, w2T, TH, TD);

  // gate MLP + routing
  gemm_kernel<0><<<dim3(TG / 128, TT / 128), 256, 0, stream>>>(
      xb, wg1T, nullptr, nullptr, nullptr, nullptr, A1, nullptr);
  route_kernel<<<TT / 8, 256, 0, stream>>>(A1, wg2, probs, top_idx, top_val, counts);
  prefix_kernel<<<1, 32, 0, stream>>>(counts, offsets, cursor);
  scatter_kernel<<<TT / 256, 256, 0, stream>>>(top_idx, top_val, offsets, cursor, tok_map, gate_map);

  // grouped expert FFN (gather -> GEMM -> gelu*gate -> GEMM -> scatter-add)
  gemm_kernel<1><<<dim3(TH / 128, TT / 128, TE), 256, 0, stream>>>(
      xb, w1T, tok_map, gate_map, counts, offsets, hb, nullptr);
  gemm_kernel<2><<<dim3(TD / 128, TT / 128, TE), 256, 0, stream>>>(
      hb, w2T, tok_map, nullptr, counts, offsets, nullptr, y);

  loss_kernel<<<1, 256, 0, stream>>>(probs, y + (size_t)(out_size - 1));
}
